// ActiveInferenceStep_87050397155586
// MI455X (gfx1250) — compile-verified, matrix operand order checked by reading, not run on hardware
//
#include <hip/hip_runtime.h>
#include <hip/hip_bf16.h>
#include <math.h>

typedef __attribute__((ext_vector_type(16))) _Float16 v16h;
typedef __attribute__((ext_vector_type(8)))  float    v8f;
typedef __attribute__((ext_vector_type(4)))  float    f32x4;

#define N_NODES 200000
#define D_FEAT  64
#define M_FAC   400000
#define K_VARS  3
#define S_ST    4
#define N_TILES 12500   // N_NODES / 16
#define NEG_INF (-1.0e30f)

// Hardware tanh (TRANS32 v_tanh_f32 on gfx1250) with safe fallback.
__device__ __forceinline__ float fast_tanh(float x) {
#if defined(__has_builtin)
#if __has_builtin(__builtin_amdgcn_tanhf)
    return __builtin_amdgcn_tanhf(x);
#elif __has_builtin(__builtin_amdgcn_tanh_f32)
    return __builtin_amdgcn_tanh_f32(x);
#else
    return tanhf(x);
#endif
#else
    return tanhf(x);
#endif
}

// ---------------------------------------------------------------- utilities
__global__ void zero_kernel(float* __restrict__ p, int n) {
    int i = blockIdx.x * blockDim.x + threadIdx.x;
    if (i < n) p[i] = 0.0f;
}

__global__ void copy_kernel(float* __restrict__ dst, const float* __restrict__ src, int n) {
    int i = blockIdx.x * blockDim.x + threadIdx.x;
    if (i < n) dst[i] = src[i];
}

// ---------------------------------------------------------------- degree
__global__ void deg_kernel(const int* __restrict__ flat, float* __restrict__ deg) {
    int i = blockIdx.x * blockDim.x + threadIdx.x;
    if (i < M_FAC * K_VARS) atomicAdd(&deg[flat[i]], 1.0f);
}

__global__ void invdeg_kernel(const float* __restrict__ deg, float* __restrict__ invd) {
    int i = blockIdx.x * blockDim.x + threadIdx.x;
    if (i < N_NODES) invd[i] = 1.0f / fmaxf(deg[i], 1.0f);
}

// ---------------------------------------------------------------- agg(h): edge mean -> scatter add
__global__ void agg_scatter_kernel(const float* __restrict__ h,
                                   const int* __restrict__ factors,
                                   float* __restrict__ node) {
    int t = blockIdx.x * blockDim.x + threadIdx.x;
    if (t >= M_FAC * 64) return;
    int m = t >> 6;
    int d = t & 63;
    int i0 = factors[m * 3 + 0];
    int i1 = factors[m * 3 + 1];
    int i2 = factors[m * 3 + 2];
    float e = (h[i0 * 64 + d] + h[i1 * 64 + d] + h[i2 * 64 + d]) * (1.0f / 3.0f);
    atomicAdd(&node[i0 * 64 + d], e);
    atomicAdd(&node[i1 * 64 + d], e);
    atomicAdd(&node[i2 * 64 + d], e);
}

// ---------------------------------------------------------------- ODE step: h += dt * tanh((h+agg)@W1) @ W2
// One wave per 16-row tile; 8 waves per block. 16 v_wmma_f32_16x16x32_f16 per wave.
// Weights are pre-swizzled into per-lane WMMA B-operand layout in LDS so each
// B-tile fetch is a single 32-byte vector LDS load.
__global__ __launch_bounds__(256)
void ode_gemm_kernel(const float* __restrict__ h,
                     const float* __restrict__ aggb,
                     const float* __restrict__ invd,
                     const float* __restrict__ w1,
                     const float* __restrict__ w2,
                     float* __restrict__ hout,
                     float dt) {
    // bswz[mat][kk*4+jj][lane] = 16 halves of the B operand for that lane
    __shared__ v16h bswz[2][8][32];
    __shared__ float zs[8][16 * 72];   // per-wave tanh(Z) tile, row stride 72 (pad)

    // ---- cooperative B-operand swizzle: 256 threads -> 8 tiles x 32 lanes
    {
        int t  = threadIdx.x;
        int tt = t >> 5;          // tile (kk*4 + jj)
        int l  = t & 31;          // lane within tile
        int kk = tt >> 2;
        int jj = tt & 3;
        int kb = (l < 16) ? 0 : 8;
        int col = jj * 16 + (l & 15);
        v16h t1, t2;
#pragma unroll
        for (int e = 0; e < 16; ++e) {
            int k = kb + ((e < 8) ? e : (16 + (e - 8)));
            int idx = (kk * 32 + k) * 64 + col;
            t1[e] = (_Float16)w1[idx];
            t2[e] = (_Float16)w2[idx];
        }
        bswz[0][tt][l] = t1;
        bswz[1][tt][l] = t2;
    }
    __syncthreads();

    const int wave = threadIdx.x >> 5;
    const int lane = threadIdx.x & 31;
    const int tile = blockIdx.x * 8 + wave;
    if (tile >= N_TILES) return;

    const int r0    = tile * 16;
    const int mrow  = lane & 15;
    const int kbase = (lane < 16) ? 0 : 8;   // 16-bit A layout: K-half per lane group
    const int col16 = lane & 15;
    const int rowb  = (r0 + mrow) * 64;
    const float idg = invd[r0 + mrow];

    // ---- A tiles (16x32 f16 each) from xin = h + agg*inv_deg, vectorized loads
    // elements 0..7  <-> k = kbase+0..7   (contiguous)
    // elements 8..15 <-> k = kbase+16..23 (contiguous)
    f32x4 hv[8], av[8];
#pragma unroll
    for (int q = 0; q < 4; ++q) {          // 4 contiguous runs of 8 floats
        int base = rowb + kbase + ((q & 1) ? 16 : 0) + ((q >> 1) ? 32 : 0);
        hv[q * 2 + 0] = *(const f32x4*)(h + base);
        hv[q * 2 + 1] = *(const f32x4*)(h + base + 4);
        av[q * 2 + 0] = *(const f32x4*)(aggb + base);
        av[q * 2 + 1] = *(const f32x4*)(aggb + base + 4);
    }
    v16h alo, ahi;
#pragma unroll
    for (int e = 0; e < 8; ++e) {
        alo[e]     = (_Float16)(hv[0 + (e >> 2)][e & 3] + av[0 + (e >> 2)][e & 3] * idg);
        alo[e + 8] = (_Float16)(hv[2 + (e >> 2)][e & 3] + av[2 + (e >> 2)][e & 3] * idg);
        ahi[e]     = (_Float16)(hv[4 + (e >> 2)][e & 3] + av[4 + (e >> 2)][e & 3] * idg);
        ahi[e + 8] = (_Float16)(hv[6 + (e >> 2)][e & 3] + av[6 + (e >> 2)][e & 3] * idg);
    }

    // ---- GEMM 1: Z = Xin @ W1  (4 col tiles x 2 K-steps)
    v8f acc[4];
#pragma unroll
    for (int jj = 0; jj < 4; ++jj) {
        v16h blo = bswz[0][jj][lane];
        v16h bhi = bswz[0][4 + jj][lane];
        v8f c = {};
        c = __builtin_amdgcn_wmma_f32_16x16x32_f16(false, alo, false, blo, (short)0, c, false, false);
        c = __builtin_amdgcn_wmma_f32_16x16x32_f16(false, ahi, false, bhi, (short)0, c, false, false);
        acc[jj] = c;
    }

    // ---- tanh (hw v_tanh_f32), C-layout -> LDS (wave-private region, no barrier)
    const int rofs = (lane >> 4) << 3;
    float* z = zs[wave];
#pragma unroll
    for (int jj = 0; jj < 4; ++jj)
#pragma unroll
        for (int v = 0; v < 8; ++v)
            z[(v + rofs) * 72 + jj * 16 + col16] = fast_tanh(acc[jj][v]);

    // ---- reload in A layout with vector LDS loads (+ f32->f16 convert)
    f32x4 zv[8];
#pragma unroll
    for (int q = 0; q < 4; ++q) {
        int base = mrow * 72 + kbase + ((q & 1) ? 16 : 0) + ((q >> 1) ? 32 : 0);
        zv[q * 2 + 0] = *(const f32x4*)(z + base);
        zv[q * 2 + 1] = *(const f32x4*)(z + base + 4);
    }
    v16h a2lo, a2hi;
#pragma unroll
    for (int e = 0; e < 8; ++e) {
        a2lo[e]     = (_Float16)zv[0 + (e >> 2)][e & 3];
        a2lo[e + 8] = (_Float16)zv[2 + (e >> 2)][e & 3];
        a2hi[e]     = (_Float16)zv[4 + (e >> 2)][e & 3];
        a2hi[e + 8] = (_Float16)zv[6 + (e >> 2)][e & 3];
    }

    // ---- GEMM 2: dh = tanh(Z) @ W2, epilogue h += dt*dh
#pragma unroll
    for (int jj = 0; jj < 4; ++jj) {
        v16h blo = bswz[1][jj][lane];
        v16h bhi = bswz[1][4 + jj][lane];
        v8f c = {};
        c = __builtin_amdgcn_wmma_f32_16x16x32_f16(false, a2lo, false, blo, (short)0, c, false, false);
        c = __builtin_amdgcn_wmma_f32_16x16x32_f16(false, a2hi, false, bhi, (short)0, c, false, false);
#pragma unroll
        for (int v = 0; v < 8; ++v) {
            int grow = (r0 + v + rofs) * 64;
            int gcol = jj * 16 + col16;
            hout[grow + gcol] = h[grow + gcol] + dt * c[v];
        }
    }
}

// ---------------------------------------------------------------- evidence = log_softmax(((x+agg)@Wconv)[:, :4])
__global__ void evidence_kernel(const float* __restrict__ h,
                                const float* __restrict__ aggb,
                                const float* __restrict__ invd,
                                const float* __restrict__ wc,
                                float* __restrict__ ev) {
    int n = blockIdx.x * blockDim.x + threadIdx.x;
    if (n >= N_NODES) return;
    float idg = invd[n];
    float l0 = 0.f, l1 = 0.f, l2 = 0.f, l3 = 0.f;
    int base = n * 64;
    for (int d = 0; d < 64; ++d) {
        float xa = h[base + d] + aggb[base + d] * idg;
        l0 += xa * wc[d * 8 + 0];
        l1 += xa * wc[d * 8 + 1];
        l2 += xa * wc[d * 8 + 2];
        l3 += xa * wc[d * 8 + 3];
    }
    float mx = fmaxf(fmaxf(l0, l1), fmaxf(l2, l3));
    float se = expf(l0 - mx) + expf(l1 - mx) + expf(l2 - mx) + expf(l3 - mx);
    float lse = mx + logf(se);
    ev[n * 4 + 0] = l0 - lse;
    ev[n * 4 + 1] = l1 - lse;
    ev[n * 4 + 2] = l2 - lse;
    ev[n * 4 + 3] = l3 - lse;
}

// ---------------------------------------------------------------- BP: node_sum scatter
__global__ void nodesum_kernel(const float* __restrict__ ftov,
                               const int* __restrict__ flat,
                               float* __restrict__ ns) {
    int i = blockIdx.x * blockDim.x + threadIdx.x;
    if (i >= M_FAC * K_VARS) return;
    int n = flat[i];
    atomicAdd(&ns[n * 4 + 0], ftov[i * 4 + 0]);
    atomicAdd(&ns[n * 4 + 1], ftov[i * 4 + 1]);
    atomicAdd(&ns[n * 4 + 2], ftov[i * 4 + 2]);
    atomicAdd(&ns[n * 4 + 3], ftov[i * 4 + 3]);
}

// ---------------------------------------------------------------- BP: per-factor max-product update
__global__ void bp_update_kernel(const int* __restrict__ factors,
                                 const float* __restrict__ lp,
                                 const float* __restrict__ ev,
                                 const float* __restrict__ ns,
                                 float* __restrict__ ftov) {
    int m = blockIdx.x * blockDim.x + threadIdx.x;
    if (m >= M_FAC) return;

    float vt[3][4], old[3][4];
#pragma unroll
    for (int k = 0; k < 3; ++k) {
        int idx = factors[m * 3 + k];
        float rmax = NEG_INF;
#pragma unroll
        for (int s = 0; s < 4; ++s) {
            float f = ftov[(m * 3 + k) * 4 + s];
            old[k][s] = f;
            float b = ev[idx * 4 + s] + ns[idx * 4 + s] - f;
            vt[k][s] = b;
            rmax = fmaxf(rmax, b);
        }
#pragma unroll
        for (int s = 0; s < 4; ++s) vt[k][s] -= rmax;
    }

    float n0[4] = {NEG_INF, NEG_INF, NEG_INF, NEG_INF};
    float n1[4] = {NEG_INF, NEG_INF, NEG_INF, NEG_INF};
    float n2[4] = {NEG_INF, NEG_INF, NEG_INF, NEG_INF};
    const float* lpr = lp + m * 64;
#pragma unroll
    for (int t = 0; t < 64; ++t) {
        int s0 = t >> 4, s1 = (t >> 2) & 3, s2 = t & 3;
        float v = lpr[t];
        n0[s0] = fmaxf(n0[s0], v + vt[1][s1] + vt[2][s2]);
        n1[s1] = fmaxf(n1[s1], v + vt[0][s0] + vt[2][s2]);
        n2[s2] = fmaxf(n2[s2], v + vt[0][s0] + vt[1][s1]);
    }

    {
        float rmax = fmaxf(fmaxf(n0[0], n0[1]), fmaxf(n0[2], n0[3]));
#pragma unroll
        for (int s = 0; s < 4; ++s)
            ftov[(m * 3 + 0) * 4 + s] = 0.5f * old[0][s] + 0.5f * (n0[s] - rmax);
    }
    {
        float rmax = fmaxf(fmaxf(n1[0], n1[1]), fmaxf(n1[2], n1[3]));
#pragma unroll
        for (int s = 0; s < 4; ++s)
            ftov[(m * 3 + 1) * 4 + s] = 0.5f * old[1][s] + 0.5f * (n1[s] - rmax);
    }
    {
        float rmax = fmaxf(fmaxf(n2[0], n2[1]), fmaxf(n2[2], n2[3]));
#pragma unroll
        for (int s = 0; s < 4; ++s)
            ftov[(m * 3 + 2) * 4 + s] = 0.5f * old[2][s] + 0.5f * (n2[s] - rmax);
    }
}

// ---------------------------------------------------------------- final marginals + argmax
__global__ void final_kernel(const float* __restrict__ ev,
                             const float* __restrict__ ns,
                             float* __restrict__ out) {
    int n = blockIdx.x * blockDim.x + threadIdx.x;
    if (n >= N_NODES) return;
    float b[4];
#pragma unroll
    for (int s = 0; s < 4; ++s) b[s] = ev[n * 4 + s] + ns[n * 4 + s];
    float mx = fmaxf(fmaxf(b[0], b[1]), fmaxf(b[2], b[3]));
    float e[4], se = 0.f;
#pragma unroll
    for (int s = 0; s < 4; ++s) { e[s] = expf(b[s] - mx); se += e[s]; }
    float inv = 1.0f / se;
#pragma unroll
    for (int s = 0; s < 4; ++s) out[n * 4 + s] = e[s] * inv;
    int am = 0;
#pragma unroll
    for (int s = 1; s < 4; ++s) if (b[s] > b[am]) am = s;
    out[N_NODES * 4 + n] = (float)am;
}

// ---------------------------------------------------------------- host orchestration
extern "C" void kernel_launch(void* const* d_in, const int* in_sizes, int n_in,
                              void* d_out, int out_size, void* d_ws, size_t ws_size,
                              hipStream_t stream) {
    const float* x       = (const float*)d_in[0];
    const int*   factors = (const int*)  d_in[1];
    const float* lp      = (const float*)d_in[2];
    const float* w1      = (const float*)d_in[3];
    const float* w2      = (const float*)d_in[4];
    const float* wc      = (const float*)d_in[5];
    float* out = (float*)d_out;

    const int ND = N_NODES * D_FEAT;       // 12.8M
    const int MK = M_FAC * K_VARS;         // 1.2M
    const int FT = M_FAC * K_VARS * S_ST;  // 4.8M

    char* ws = (char*)d_ws;
    size_t off = 0;
    auto take = [&](size_t bytes) -> char* {
        char* p = ws + off;
        off += (bytes + 255) & ~(size_t)255;
        return p;
    };
    float* h0   = (float*)take((size_t)ND * 4);
    float* h1   = (float*)take((size_t)ND * 4);
    float* agg  = (float*)take((size_t)ND * 4);
    float* deg  = (float*)take((size_t)N_NODES * 4);
    float* invd = (float*)take((size_t)N_NODES * 4);
    float* ev   = (float*)take((size_t)N_NODES * 4 * 4);
    float* ns   = (float*)take((size_t)N_NODES * 4 * 4);
    float* ftov = (float*)take((size_t)FT * 4);

    const int B = 256;
    dim3 blk(B);
    auto grid1 = [&](int n) { return dim3((unsigned)((n + B - 1) / B)); };

    // init: h0 = x, degrees
    copy_kernel<<<grid1(ND), blk, 0, stream>>>(h0, x, ND);
    zero_kernel<<<grid1(N_NODES), blk, 0, stream>>>(deg, N_NODES);
    deg_kernel<<<grid1(MK), blk, 0, stream>>>(factors, deg);
    invdeg_kernel<<<grid1(N_NODES), blk, 0, stream>>>(deg, invd);

    // ODE: 4 steps, ping-pong h0/h1
    float* hc = h0;
    float* hn = h1;
    const int aggN = M_FAC * 64;
    dim3 ggrid((unsigned)((N_TILES + 7) / 8));
    for (int t = 0; t < 4; ++t) {
        zero_kernel<<<grid1(ND), blk, 0, stream>>>(agg, ND);
        agg_scatter_kernel<<<grid1(aggN), blk, 0, stream>>>(hc, factors, agg);
        ode_gemm_kernel<<<ggrid, blk, 0, stream>>>(hc, agg, invd, w1, w2, hn, 0.25f);
        float* tmp = hc; hc = hn; hn = tmp;
    }

    // evidence
    zero_kernel<<<grid1(ND), blk, 0, stream>>>(agg, ND);
    agg_scatter_kernel<<<grid1(aggN), blk, 0, stream>>>(hc, factors, agg);
    evidence_kernel<<<grid1(N_NODES), blk, 0, stream>>>(hc, agg, invd, wc, ev);

    // BP
    zero_kernel<<<grid1(FT), blk, 0, stream>>>(ftov, FT);
    for (int it = 0; it < 16; ++it) {
        zero_kernel<<<grid1(N_NODES * 4), blk, 0, stream>>>(ns, N_NODES * 4);
        nodesum_kernel<<<grid1(MK), blk, 0, stream>>>(ftov, factors, ns);
        bp_update_kernel<<<grid1(M_FAC), blk, 0, stream>>>(factors, lp, ev, ns, ftov);
    }

    // final belief -> marginals + argmax
    zero_kernel<<<grid1(N_NODES * 4), blk, 0, stream>>>(ns, N_NODES * 4);
    nodesum_kernel<<<grid1(MK), blk, 0, stream>>>(ftov, factors, ns);
    final_kernel<<<grid1(N_NODES), blk, 0, stream>>>(ev, ns, out);
}
